// DecoderStory_6425271075135
// MI455X (gfx1250) — matrix operand order repaired and checked by reading, not compiled, not run on hardware
//
#include <hip/hip_runtime.h>
#include <hip/hip_bf16.h>

// ---------------------------------------------------------------------------
// MI455X (gfx1250) implementation of the 2-layer LSTM story decoder.
//   * bf16 weight copies (halve L2 traffic; all weights ~150MB -> L2 resident)
//   * WMMA f32_16x16x32_bf16 for: feat GEMM, input-projection GEMMs (hoisted
//     out of the sequential recurrence), recurrent h@Whh^T GEMVs, logits GEMM
//   * GEMM: 16x64 tile per wave (4 accumulators, A-fragment reuse x4)
//   * persistent 32-WG cooperative kernel with atomic grid barrier for the
//     strictly sequential 500-step recurrence; h vectors staged in LDS
// ---------------------------------------------------------------------------

typedef __bf16 bf16;
typedef __attribute__((ext_vector_type(16))) __bf16 v16bf;
typedef __attribute__((ext_vector_type(8)))  __bf16 v8bf;
typedef __attribute__((ext_vector_type(8)))  float  v8f;

#define NWG   32
#define BLK   256
#define HDIM  1024
#define GDIM  4096     // 4*H gates
#define EDIM  512
#define XDIM  1536     // E + H
#define VDIM  12000
#define TCAP  40
#define TDESC 60
#define BS    5

// ---------------- WMMA fragment loaders (per CDNA5 ISA) ----------

// A-matrix 16x32 bf16: lane holds row (lane&15); this code's contiguous K order 16*(lane>>4)..+15 (B matches).
__device__ __forceinline__ v16bf load_a_frag(const bf16* rowp, int k0) {
  v8bf lo = *(const v8bf*)(rowp + k0);
  v8bf hi = *(const v8bf*)(rowp + k0 + 8);
  v16bf r;
#pragma unroll
  for (int i = 0; i < 8; ++i) { r[i] = lo[i]; r[i + 8] = hi[i]; }
  return r;
}

#define WMMA_BF16(acc, a, b)                                                  \
  (__builtin_amdgcn_wmma_f32_16x16x32_bf16(false, (a), false, (b), (short)0,  \
                                           (acc), false, false))

__device__ __forceinline__ float sigf(float x) { return 1.0f / (1.0f + __expf(-x)); }

// ---------------- generic WMMA GEMM:  C[M,N] = A[M,K] @ W[N,K]^T (+bias) ----
// one wave per 16x64 output tile: 4 accumulators share one A fragment per
// k-step (4x A reuse).  grid = (ceil(N/64), ceil(M/16))
__global__ void __launch_bounds__(32) k_gemm_bf16(
    float* __restrict__ C, const bf16* __restrict__ A, const bf16* __restrict__ W,
    const float* __restrict__ bias, int M, int N, int K, int relu) {
  int lane = threadIdx.x;
  int half = lane >> 4, l16 = lane & 15;
  int mt = blockIdx.y;
  int ntiles = N >> 4;
  int nt0 = blockIdx.x * 4;
  int arow = mt * 16 + l16;
  const bf16* ap = A + (size_t)(arow < M ? arow : 0) * K + half * 16;
  // B-matrix 32x16 bf16: lane holds col (lane&15), contiguous K 16*(lane>>4)..+15, same order as A
  const bf16* wp0;
  const bf16* wp1;
  const bf16* wp2;
  const bf16* wp3;
  {
    int t0 = nt0 + 0 < ntiles ? nt0 + 0 : ntiles - 1;
    int t1 = nt0 + 1 < ntiles ? nt0 + 1 : ntiles - 1;
    int t2 = nt0 + 2 < ntiles ? nt0 + 2 : ntiles - 1;
    int t3 = nt0 + 3 < ntiles ? nt0 + 3 : ntiles - 1;
    wp0 = W + (size_t)(t0 * 16 + l16) * K + half * 16;
    wp1 = W + (size_t)(t1 * 16 + l16) * K + half * 16;
    wp2 = W + (size_t)(t2 * 16 + l16) * K + half * 16;
    wp3 = W + (size_t)(t3 * 16 + l16) * K + half * 16;
  }
  v8f acc0 = {}, acc1 = {}, acc2 = {}, acc3 = {};
  for (int k0 = 0; k0 < K; k0 += 32) {
    v16bf a  = load_a_frag(ap, k0);
    v16bf b0 = *(const v16bf*)(wp0 + k0);
    v16bf b1 = *(const v16bf*)(wp1 + k0);
    v16bf b2 = *(const v16bf*)(wp2 + k0);
    v16bf b3 = *(const v16bf*)(wp3 + k0);
    acc0 = WMMA_BF16(acc0, a, b0);
    acc1 = WMMA_BF16(acc1, a, b1);
    acc2 = WMMA_BF16(acc2, a, b2);
    acc3 = WMMA_BF16(acc3, a, b3);
  }
  v8f accs[4] = {acc0, acc1, acc2, acc3};
#pragma unroll
  for (int j = 0; j < 4; ++j) {
    int nt = nt0 + j;
    if (nt >= ntiles) break;
    int n = nt * 16 + l16;
    float bv = bias ? bias[n] : 0.0f;
#pragma unroll
    for (int v = 0; v < 8; ++v) {
      int m = mt * 16 + v + half * 8;   // C layout: M = vgpr + 8*(lane>>4)
      if (m < M) {
        float r = accs[j][v] + bv;
        if (relu) r = fmaxf(r, 0.0f);
        C[(size_t)m * N + n] = r;
      }
    }
  }
}

// ---------------- recurrent GEMV tile via WMMA ------------------------------
// acc += Whh[row0..row0+15, 0..H-1] @ h[0..H-1]; h comes from LDS, broadcast
// into all 16 B columns so every C column carries the result.
__device__ __forceinline__ void gemv_tile(v8f& acc, const bf16* W, int row0,
                                          const bf16* hlds, int lane) {
  int half = lane >> 4, l16 = lane & 15;
  const bf16* wp = W + (size_t)(row0 + l16) * HDIM + half * 16;
  const bf16* hp = hlds + half * 16;
  for (int k0 = 0; k0 < HDIM; k0 += 32) {
    v16bf a = load_a_frag(wp, k0);
    v16bf b = *(const v16bf*)(hp + k0);     // LDS read (same chunk per column)
    acc = WMMA_BF16(acc, a, b);
  }
}

// cooperative copy of 1024 bf16 (2KB) global -> LDS: 128 threads x 16B
__device__ __forceinline__ void stage_h(bf16* dst, const bf16* src, int tid) {
  if (tid < 128) *(uint4*)(dst + tid * 8) = *(const uint4*)(src + tid * 8);
}

// ---------------- grid-wide barrier (persistent cooperative kernel) ---------
struct Sync { unsigned* bar; unsigned count; };

__device__ __forceinline__ void gsync(Sync& s) {
  __threadfence();
  __syncthreads();
  if (threadIdx.x == 0) {
    __hip_atomic_fetch_add(s.bar, 1u, __ATOMIC_RELEASE, __HIP_MEMORY_SCOPE_AGENT);
    unsigned target = (s.count + 1u) * NWG;
    while (__hip_atomic_load(s.bar, __ATOMIC_ACQUIRE, __HIP_MEMORY_SCOPE_AGENT) < target)
      __builtin_amdgcn_s_sleep(8);
  }
  s.count++;
  __syncthreads();
  __threadfence();
}

// ---------------- persistent LSTM kernel ------------------------------------
struct PParams {
  const bf16 *mWhh0, *mWih1, *mWhh1;
  const bf16 *dWhh0, *dWih1, *dWhh1;
  const float *xm, *xd;                 // precomputed x@Wih0^T + (bih0+bhh0)
  const float *m_bih1, *m_bhh1, *d_bih1, *d_bhh1;
  const float *hidW, *hidb, *hid2W, *hid2b, *feat;
  const int *lengths, *desc_lengths;
  float *Sd, *Sm, *tmp, *ysf;           // S layout: [h0,c0,h1,c1] x 1024
  bf16 *hb;                             // [par0 h0|par0 h1|par1 h0|par1 h1|h0raw]
  unsigned *bar;
};

__device__ __forceinline__ float catv(const float* Sd, const float* Sm, int isC, int k) {
  const float* S = (k < 2048) ? Sd : Sm;
  int kk = k & 2047;
  return S[(kk >> 10) * 2048 + isC * HDIM + (kk & (HDIM - 1))];
}

__device__ void run_phase(const bf16* Whh0, const bf16* Wih1, const bf16* Whh1,
                          const float* xproj, const float* bih1, const float* bhh1,
                          float* S, bf16* hb, float* ys, int T, int valid,
                          float* gbuf, bf16* sh, int wg, int tid, Sync& sc) {
  int lane = tid & 31, wv = tid >> 5;                         // 8 waves
  int grow  = (wv >> 1) * HDIM + wg * 32 + (wv & 1) * 16;     // gate-row tile base
  int lbase = (wv >> 1) * 32 + (wv & 1) * 16;
  for (int t = 0; t < T; ++t) {
    int cur = (t & 1) * 2048, nxt = 2048 - cur;
    // ----- layer 0: gates = xproj(t) + Whh0 @ h0 -----
    stage_h(sh, hb + cur, tid);                    // h0 carry -> LDS
    __syncthreads();
    v8f acc = {};
    gemv_tile(acc, Whh0, grow, sh, lane);
    if ((lane & 15) == 0) {
      int mb = (lane >> 4) * 8;
#pragma unroll
      for (int v = 0; v < 8; ++v) gbuf[lbase + mb + v] = acc[v];
    }
    __syncthreads();
    if (tid < 32) {
      int j = wg * 32 + tid;
      const float* xp = xproj + (size_t)t * GDIM;
      float gi = xp[j]            + gbuf[tid];
      float gf = xp[HDIM + j]     + gbuf[32 + tid];
      float gg = xp[2 * HDIM + j] + gbuf[64 + tid];
      float go = xp[3 * HDIM + j] + gbuf[96 + tid];
      float cold = S[HDIM + j], hold = S[j];
      float cn = sigf(gf) * cold + sigf(gi) * tanhf(gg);
      float hn = sigf(go) * tanhf(cn);
      hb[4096 + j] = (bf16)hn;                       // raw output feeds layer 1
      float hk = (t < valid) ? hn : hold;
      float ck = (t < valid) ? cn : cold;
      S[j] = hk; S[HDIM + j] = ck;
      hb[nxt + j] = (bf16)hk;                        // double-buffered carry
    }
    gsync(sc);
    // ----- layer 1: gates = Wih1 @ h0raw + Whh1 @ h1 + b -----
    stage_h(sh + HDIM, hb + 4096, tid);            // raw layer-0 output -> LDS
    stage_h(sh + 2 * HDIM, hb + cur + HDIM, tid);  // h1 carry -> LDS
    __syncthreads();
    v8f acc2 = {};
    gemv_tile(acc2, Wih1, grow, sh + HDIM, lane);
    gemv_tile(acc2, Whh1, grow, sh + 2 * HDIM, lane);
    if ((lane & 15) == 0) {
      int mb = (lane >> 4) * 8;
#pragma unroll
      for (int v = 0; v < 8; ++v) gbuf[lbase + mb + v] = acc2[v];
    }
    __syncthreads();
    if (tid < 32) {
      int j = wg * 32 + tid;
      float gi = gbuf[tid]      + bih1[j]            + bhh1[j];
      float gf = gbuf[32 + tid] + bih1[HDIM + j]     + bhh1[HDIM + j];
      float gg = gbuf[64 + tid] + bih1[2*HDIM + j]   + bhh1[2*HDIM + j];
      float go = gbuf[96 + tid] + bih1[3*HDIM + j]   + bhh1[3*HDIM + j];
      float cold = S[3*HDIM + j], hold = S[2*HDIM + j];
      float cn = sigf(gf) * cold + sigf(gi) * tanhf(gg);
      float hn = sigf(go) * tanhf(cn);
      if (ys) ys[(size_t)t * HDIM + j] = hn;         // raw output
      float hk = (t < valid) ? hn : hold;
      float ck = (t < valid) ? cn : cold;
      S[2*HDIM + j] = hk; S[3*HDIM + j] = ck;
      hb[nxt + HDIM + j] = (bf16)hk;
    }
    gsync(sc);
  }
}

__global__ void __launch_bounds__(BLK) k_lstm(PParams p) {
  __shared__ float gbuf[128];
  __shared__ __align__(32) bf16 sh[3 * HDIM];
  int tid = threadIdx.x, wg = blockIdx.x;
  int gtid = wg * BLK + tid;
  Sync sc{p.bar, 0};

  // dh = dc = hid2_W @ feat_flat + hid2_b ; main carry = 0
  if (gtid < 2048) {
    const float* wrow = p.hid2W + (size_t)gtid * (BS * HDIM);
    float acc = p.hid2b[gtid];
    for (int k = 0; k < BS * HDIM; ++k) acc += wrow[k] * p.feat[k];
    int layer = gtid >> 10, j = gtid & (HDIM - 1);
    p.Sd[layer * 2048 + j] = acc;
    p.Sd[layer * 2048 + HDIM + j] = acc;
  }
  if (gtid < 4096) p.Sm[gtid] = 0.0f;
  gsync(sc);

  for (int story = 0; story < BS; ++story) {
    if (gtid < HDIM) {                                 // seed parity-0 h buffers
      p.hb[gtid]        = (bf16)p.Sd[gtid];
      p.hb[HDIM + gtid] = (bf16)p.Sd[2048 + gtid];
    }
    gsync(sc);
    run_phase(p.dWhh0, p.dWih1, p.dWhh1, p.xd + (size_t)story * TDESC * GDIM,
              p.d_bih1, p.d_bhh1, p.Sd, p.hb, nullptr,
              TDESC, p.desc_lengths[story] - 1, gbuf, sh, wg, tid, sc);
    // hidden transform: tmp[0..2047]=h', tmp[2048..4095]=c'
    if (gtid < 4096) {
      int isC = gtid >> 11, r = gtid & 2047;
      const float* wrow = p.hidW + (size_t)r * GDIM;
      float acc = p.hidb[r];
      for (int k = 0; k < GDIM; ++k) acc += wrow[k] * catv(p.Sd, p.Sm, isC, k);
      p.tmp[gtid] = acc;
    }
    gsync(sc);
    if (gtid < HDIM) {
      p.Sm[gtid]          = p.tmp[gtid];
      p.Sm[2048 + gtid]   = p.tmp[HDIM + gtid];
      p.Sm[HDIM + gtid]   = p.tmp[2048 + gtid];
      p.Sm[3*HDIM + gtid] = p.tmp[2048 + HDIM + gtid];
      p.hb[gtid]        = (bf16)p.tmp[gtid];
      p.hb[HDIM + gtid] = (bf16)p.tmp[HDIM + gtid];
    }
    gsync(sc);
    run_phase(p.mWhh0, p.mWih1, p.mWhh1, p.xm + (size_t)story * TCAP * GDIM,
              p.m_bih1, p.m_bhh1, p.Sm, p.hb, p.ysf + (size_t)story * TCAP * HDIM,
              TCAP, p.lengths[story] - 1, gbuf, sh, wg, tid, sc);
  }
}

// ---------------- small helper kernels --------------------------------------
__global__ void k_cvt(bf16* __restrict__ dst, const float* __restrict__ src, int n) {
  for (int i = blockIdx.x * blockDim.x + threadIdx.x; i < n; i += gridDim.x * blockDim.x)
    dst[i] = (bf16)src[i];
}
__global__ void k_addv(float* __restrict__ d, const float* __restrict__ a,
                       const float* __restrict__ b, int n) {
  for (int i = blockIdx.x * blockDim.x + threadIdx.x; i < n; i += gridDim.x * blockDim.x)
    d[i] = a[i] + b[i];
}
__global__ void k_zero_u32(unsigned* p, int n) {
  for (int i = blockIdx.x * blockDim.x + threadIdx.x; i < n; i += gridDim.x * blockDim.x)
    p[i] = 0u;
}
__global__ void k_build_emb(bf16* __restrict__ out, const float* __restrict__ feat,
                            const float* __restrict__ embW, const int* __restrict__ toks,
                            int T, int rows) {
  int n = rows * XDIM;
  for (int i = blockIdx.x * blockDim.x + threadIdx.x; i < n; i += gridDim.x * blockDim.x) {
    int r = i / XDIM, c = i - r * XDIM;
    int story = r / T;
    float v = (c < HDIM) ? feat[story * HDIM + c]
                         : embW[(size_t)toks[r] * EDIM + (c - HDIM)];
    out[i] = (bf16)v;
  }
}
__global__ void k_finalize(float* __restrict__ out, const float* __restrict__ logits,
                           const float* __restrict__ outb, const int* __restrict__ lens) {
  const int n = BS * TCAP * VDIM;
  for (int i = blockIdx.x * blockDim.x + threadIdx.x; i < n; i += gridDim.x * blockDim.x) {
    int story = i / (TCAP * VDIM);
    int rem = i - story * TCAP * VDIM;
    int t = rem / VDIM, v = rem - t * VDIM;
    float val;
    if (t == 0) val = (v == 1) ? 10000.0f : 0.0f;
    else        val = logits[(size_t)(story * TCAP + t - 1) * VDIM + v] + outb[v];
    if (t >= lens[story]) val = 0.0f;
    out[i] = val;
  }
}

// ---------------- host-side orchestration -----------------------------------
extern "C" void kernel_launch(void* const* d_in, const int* in_sizes, int n_in,
                              void* d_out, int out_size, void* d_ws, size_t ws_size,
                              hipStream_t stream) {
  (void)in_sizes; (void)n_in; (void)out_size; (void)ws_size;
  const float* sfeat  = (const float*)d_in[0];
  const float* embW   = (const float*)d_in[1];
  const float* storyW = (const float*)d_in[2];
  const float* storyb = (const float*)d_in[3];
  const float* outW   = (const float*)d_in[4];
  const float* outb   = (const float*)d_in[5];
  const float* hidW   = (const float*)d_in[6];
  const float* hidb   = (const float*)d_in[7];
  const float* hid2W  = (const float*)d_in[8];
  const float* hid2b  = (const float*)d_in[9];
  const float* mWih0  = (const float*)d_in[10];
  const float* mWhh0f = (const float*)d_in[11];
  const float* mbih0  = (const float*)d_in[12];
  const float* mbhh0  = (const float*)d_in[13];
  const float* mWih1f = (const float*)d_in[14];
  const float* mWhh1f = (const float*)d_in[15];
  const float* mbih1  = (const float*)d_in[16];
  const float* mbhh1  = (const float*)d_in[17];
  const float* dWih0  = (const float*)d_in[18];
  const float* dWhh0f = (const float*)d_in[19];
  const float* dbih0  = (const float*)d_in[20];
  const float* dbhh0  = (const float*)d_in[21];
  const float* dWih1f = (const float*)d_in[22];
  const float* dWhh1f = (const float*)d_in[23];
  const float* dbih1  = (const float*)d_in[24];
  const float* dbhh1  = (const float*)d_in[25];
  const int* captions = (const int*)d_in[26];
  const int* lengths  = (const int*)d_in[27];
  const int* descs    = (const int*)d_in[28];
  const int* desclens = (const int*)d_in[29];
  float* out = (float*)d_out;

  char* ws = (char*)d_ws;
  size_t off = 0;
  auto carve = [&](size_t bytes) -> void* {
    void* p = ws + off;
    off = (off + bytes + 255) & ~(size_t)255;
    return p;
  };
  bf16* b_mWih0 = (bf16*)carve(sizeof(bf16) * (size_t)GDIM * XDIM);
  bf16* b_mWhh0 = (bf16*)carve(sizeof(bf16) * (size_t)GDIM * HDIM);
  bf16* b_mWih1 = (bf16*)carve(sizeof(bf16) * (size_t)GDIM * HDIM);
  bf16* b_mWhh1 = (bf16*)carve(sizeof(bf16) * (size_t)GDIM * HDIM);
  bf16* b_dWih0 = (bf16*)carve(sizeof(bf16) * (size_t)GDIM * XDIM);
  bf16* b_dWhh0 = (bf16*)carve(sizeof(bf16) * (size_t)GDIM * HDIM);
  bf16* b_dWih1 = (bf16*)carve(sizeof(bf16) * (size_t)GDIM * HDIM);
  bf16* b_dWhh1 = (bf16*)carve(sizeof(bf16) * (size_t)GDIM * HDIM);
  bf16* b_outW  = (bf16*)carve(sizeof(bf16) * (size_t)VDIM * HDIM);
  bf16* b_stW   = (bf16*)carve(sizeof(bf16) * (size_t)HDIM * 2048);
  bf16* b_sfeat = (bf16*)carve(sizeof(bf16) * (size_t)BS * 2048);
  bf16* b_emb   = (bf16*)carve(sizeof(bf16) * (size_t)BS * TCAP * XDIM);
  bf16* b_demb  = (bf16*)carve(sizeof(bf16) * (size_t)BS * TDESC * XDIM);
  bf16* b_ys    = (bf16*)carve(sizeof(bf16) * (size_t)BS * TCAP * HDIM);
  bf16* b_hb    = (bf16*)carve(sizeof(bf16) * (size_t)5 * HDIM);
  float* f_feat = (float*)carve(sizeof(float) * (size_t)BS * HDIM);
  float* f_xm   = (float*)carve(sizeof(float) * (size_t)BS * TCAP * GDIM);
  float* f_xd   = (float*)carve(sizeof(float) * (size_t)BS * TDESC * GDIM);
  float* f_ys   = (float*)carve(sizeof(float) * (size_t)BS * TCAP * HDIM);
  float* f_log  = (float*)carve(sizeof(float) * (size_t)BS * TCAP * VDIM);
  float* f_Sd   = (float*)carve(sizeof(float) * 4 * HDIM);
  float* f_Sm   = (float*)carve(sizeof(float) * 4 * HDIM);
  float* f_tmp  = (float*)carve(sizeof(float) * 4096);
  float* f_b0m  = (float*)carve(sizeof(float) * GDIM);
  float* f_b0d  = (float*)carve(sizeof(float) * GDIM);
  unsigned* bar = (unsigned*)carve(sizeof(unsigned) * 64);

  auto cvt = [&](bf16* dst, const float* src, size_t n) {
    int blocks = (int)((n + 255) / 256);
    if (blocks > 16384) blocks = 16384;
    k_cvt<<<blocks, 256, 0, stream>>>(dst, src, (int)n);
  };

  k_zero_u32<<<1, 64, 0, stream>>>(bar, 64);

  cvt(b_mWih0, mWih0,  (size_t)GDIM * XDIM);
  cvt(b_mWhh0, mWhh0f, (size_t)GDIM * HDIM);
  cvt(b_mWih1, mWih1f, (size_t)GDIM * HDIM);
  cvt(b_mWhh1, mWhh1f, (size_t)GDIM * HDIM);
  cvt(b_dWih0, dWih0,  (size_t)GDIM * XDIM);
  cvt(b_dWhh0, dWhh0f, (size_t)GDIM * HDIM);
  cvt(b_dWih1, dWih1f, (size_t)GDIM * HDIM);
  cvt(b_dWhh1, dWhh1f, (size_t)GDIM * HDIM);
  cvt(b_outW,  outW,   (size_t)VDIM * HDIM);
  cvt(b_stW,   storyW, (size_t)HDIM * 2048);
  cvt(b_sfeat, sfeat,  (size_t)BS * 2048);

  k_addv<<<16, 256, 0, stream>>>(f_b0m, mbih0, mbhh0, GDIM);
  k_addv<<<16, 256, 0, stream>>>(f_b0d, dbih0, dbhh0, GDIM);

  // feat = relu(story_feature @ story_W^T + story_b)   [5 x 1024]
  k_gemm_bf16<<<dim3((HDIM / 16 + 3) / 4, 1), 32, 0, stream>>>(
      f_feat, b_sfeat, b_stW, storyb, BS, HDIM, 2048, 1);

  // build bf16 [feat | embedding] input rows
  k_build_emb<<<1200, 256, 0, stream>>>(b_emb,  f_feat, embW, captions, TCAP,  BS * TCAP);
  k_build_emb<<<1800, 256, 0, stream>>>(b_demb, f_feat, embW, descs,    TDESC, BS * TDESC);

  // hoisted input projections (include bih0+bhh0)
  k_gemm_bf16<<<dim3((GDIM / 16 + 3) / 4, (BS * TCAP + 15) / 16), 32, 0, stream>>>(
      f_xm, b_emb, b_mWih0, f_b0m, BS * TCAP, GDIM, XDIM, 0);
  k_gemm_bf16<<<dim3((GDIM / 16 + 3) / 4, (BS * TDESC + 15) / 16), 32, 0, stream>>>(
      f_xd, b_demb, b_dWih0, f_b0d, BS * TDESC, GDIM, XDIM, 0);

  // sequential recurrence (persistent cooperative kernel)
  PParams pp;
  pp.mWhh0 = b_mWhh0; pp.mWih1 = b_mWih1; pp.mWhh1 = b_mWhh1;
  pp.dWhh0 = b_dWhh0; pp.dWih1 = b_dWih1; pp.dWhh1 = b_dWhh1;
  pp.xm = f_xm; pp.xd = f_xd;
  pp.m_bih1 = mbih1; pp.m_bhh1 = mbhh1; pp.d_bih1 = dbih1; pp.d_bhh1 = dbhh1;
  pp.hidW = hidW; pp.hidb = hidb; pp.hid2W = hid2W; pp.hid2b = hid2b; pp.feat = f_feat;
  pp.lengths = lengths; pp.desc_lengths = desclens;
  pp.Sd = f_Sd; pp.Sm = f_Sm; pp.tmp = f_tmp; pp.ysf = f_ys;
  pp.hb = b_hb; pp.bar = bar;
  k_lstm<<<NWG, BLK, 0, stream>>>(pp);

  // logits = ys @ out_W^T  (bias folded into finalize)
  cvt(b_ys, f_ys, (size_t)BS * TCAP * HDIM);
  k_gemm_bf16<<<dim3((VDIM / 16 + 3) / 4, (BS * TCAP + 15) / 16), 32, 0, stream>>>(
      f_log, b_ys, b_outW, nullptr, BS * TCAP, VDIM, HDIM, 0);

  // assemble [B, Tc, V]: start row, shift, +out_b, zero padded rows
  k_finalize<<<4096, 256, 0, stream>>>(out, f_log, outb, lengths);
}
